// MultiHeadSelfAttention_73710228734444
// MI455X (gfx1250) — compile-verified, matrix operand order checked by reading, not run on hardware
//
#include <hip/hip_runtime.h>

// ---------------------------------------------------------------------------
// MHSA for MI455X (gfx1250): bf16 WMMA everywhere, flash-attention inner loop.
// v2: 2x M-groups per wave (better frag reuse), register double-buffered GEMM.
// ---------------------------------------------------------------------------

typedef __attribute__((ext_vector_type(16))) __bf16          v16bf;
typedef __attribute__((ext_vector_type(16))) unsigned short  v16u;
typedef __attribute__((ext_vector_type(8)))  unsigned short  v8u;
typedef __attribute__((ext_vector_type(8)))  float           v8f;

#define DIMC   768
#define NHEAD  12
#define HD     64
#define BB     2
#define NN     4096
#define MTOK   (BB * NN)        // 8192 token rows
#define QSCALE 0.125f           // 64^-0.5

union FragU { v16u u; v16bf bf; };

static __device__ __forceinline__ unsigned short f2bf(float f) {
  unsigned int u = __float_as_uint(f);
  u += 0x7FFFu + ((u >> 16) & 1u);           // round-to-nearest-even
  return (unsigned short)(u >> 16);
}

// Load one 16-bit WMMA A/B fragment (16 bf16 per lane) from a row-major row.
// Lane layout (ISA 16-bit 16x32 A-matrix): elems 0..7 -> K = kbase+8*hi+e,
// elems 8..15 -> K = kbase+16+8*hi+(e-8).  Two contiguous 16B loads.
static __device__ __forceinline__ v16u load_frag(const unsigned short* __restrict__ row,
                                                 int kbase, int hi) {
  v8u lo = *(const v8u*)(row + kbase + 8 * hi);
  v8u hc = *(const v8u*)(row + kbase + 16 + 8 * hi);
  return __builtin_shufflevector(lo, hc, 0, 1, 2, 3, 4, 5, 6, 7,
                                 8, 9, 10, 11, 12, 13, 14, 15);
}

static __device__ __forceinline__ v8f wmma_bf16(v16u a, v16u b, v8f c) {
  FragU A; A.u = a;
  FragU Bv; Bv.u = b;
  return __builtin_amdgcn_wmma_f32_16x16x32_bf16(false, A.bf, false, Bv.bf,
                                                 (short)0, c, false, false);
}

// ---------------------------------------------------------------------------
__global__ void f32_to_bf16_kernel(const float* __restrict__ in,
                                   unsigned short* __restrict__ out, int n) {
  int i = blockIdx.x * blockDim.x + threadIdx.x;
  if (i < n) out[i] = f2bf(in[i]);
}

// ---------------------------------------------------------------------------
// QKV projection: [8192x768] @ [768x2304 (=W^T)] + bias, routed to
// Q [B,H,N,64] (pre-scaled), K [B,H,N,64], Vt [B,H,64,N]; all bf16.
// Each wave: 32(M) x 64(N) tile (two 16-row groups sharing B frags);
// 4 waves along M -> 128 rows/block.  B frags double-buffered in registers.
__global__ __launch_bounds__(128) void qkv_gemm_kernel(
    const unsigned short* __restrict__ xb,   // [8192 x 768]
    const unsigned short* __restrict__ wb,   // [2304 x 768] (rows = out feat)
    const float* __restrict__ bias,          // [2304]
    unsigned short* __restrict__ Qb,
    unsigned short* __restrict__ Kb,
    unsigned short* __restrict__ Vt) {
  const int lane = threadIdx.x & 31;
  const int wave = threadIdx.x >> 5;
  const int n16  = lane & 15;
  const int hi   = lane >> 4;
  const int m0   = blockIdx.x * 128 + wave * 32;  // token tile base (A group)
  const int c0   = blockIdx.y * 64;               // out-feature base

  const unsigned short* arowA = xb + (size_t)(m0 + n16) * DIMC;
  const unsigned short* arowB = xb + (size_t)(m0 + 16 + n16) * DIMC;
  const unsigned short* brow0 = wb + (size_t)(c0 + 0 * 16 + n16) * DIMC;
  const unsigned short* brow1 = wb + (size_t)(c0 + 1 * 16 + n16) * DIMC;
  const unsigned short* brow2 = wb + (size_t)(c0 + 2 * 16 + n16) * DIMC;
  const unsigned short* brow3 = wb + (size_t)(c0 + 3 * 16 + n16) * DIMC;

  v8f ac[2][4] = {};

  // software pipeline: prefetch k+32 fragments while doing wmma on k
  v16u aA = load_frag(arowA, 0, hi), aB = load_frag(arowB, 0, hi);
  v16u b0 = load_frag(brow0, 0, hi), b1 = load_frag(brow1, 0, hi);
  v16u b2 = load_frag(brow2, 0, hi), b3 = load_frag(brow3, 0, hi);

  for (int k = 0; k < DIMC; k += 32) {
    const int kn = (k + 32 < DIMC) ? k + 32 : 0;
    v16u aAn = load_frag(arowA, kn, hi), aBn = load_frag(arowB, kn, hi);
    v16u b0n = load_frag(brow0, kn, hi), b1n = load_frag(brow1, kn, hi);
    v16u b2n = load_frag(brow2, kn, hi), b3n = load_frag(brow3, kn, hi);

    ac[0][0] = wmma_bf16(aA, b0, ac[0][0]);
    ac[0][1] = wmma_bf16(aA, b1, ac[0][1]);
    ac[0][2] = wmma_bf16(aA, b2, ac[0][2]);
    ac[0][3] = wmma_bf16(aA, b3, ac[0][3]);
    ac[1][0] = wmma_bf16(aB, b0, ac[1][0]);
    ac[1][1] = wmma_bf16(aB, b1, ac[1][1]);
    ac[1][2] = wmma_bf16(aB, b2, ac[1][2]);
    ac[1][3] = wmma_bf16(aB, b3, ac[1][3]);

    aA = aAn; aB = aBn;
    b0 = b0n; b1 = b1n; b2 = b2n; b3 = b3n;
  }

#pragma unroll
  for (int g = 0; g < 2; ++g) {
#pragma unroll
    for (int j = 0; j < 4; ++j) {
      const v8f a = ac[g][j];
      const int cb  = c0 + j * 16;        // frag column base (mult of 16)
      const int col = cb + n16;
      const float bia = bias[col];
      const int s = cb / DIMC;            // 0=Q 1=K 2=V (uniform per frag)
      const int h = (cb % DIMC) / HD;     // uniform per frag
      const int d = (cb % HD) + n16;
#pragma unroll
      for (int r = 0; r < 8; ++r) {
        const int row  = m0 + g * 16 + r + 8 * hi;   // token row
        const int bidx = row >> 12;
        const int tok  = row & (NN - 1);
        const float v = a[r] + bia;
        if (s == 0)
          Qb[((size_t)(bidx * NHEAD + h) * NN + tok) * HD + d] = f2bf(v * QSCALE);
        else if (s == 1)
          Kb[((size_t)(bidx * NHEAD + h) * NN + tok) * HD + d] = f2bf(v);
        else
          Vt[((size_t)(bidx * NHEAD + h) * HD + d) * NN + tok] = f2bf(v);
      }
    }
  }
}

// ---------------------------------------------------------------------------
// Flash attention.  Each wave owns 32 query rows (two 16-row groups sharing
// the K and V fragments), full 64-d output strip.
// Per 32-key block:  S^T tiles = K * Q^T  (C frag: lane axis = query),
// then the exp'ed C frags ARE the B frag of P^T for  O^T = V^T * P^T.
__global__ __launch_bounds__(128) void attn_kernel(
    const unsigned short* __restrict__ Qb,   // [B,H,N,64] pre-scaled bf16
    const unsigned short* __restrict__ Kb,   // [B,H,N,64] bf16
    const unsigned short* __restrict__ Vt,   // [B,H,64,N] bf16
    unsigned short* __restrict__ Ob) {       // [B,N,768]  bf16
  const int lane = threadIdx.x & 31;
  const int wave = threadIdx.x >> 5;
  const int n16  = lane & 15;
  const int hi   = lane >> 4;
  const int bh   = blockIdx.y;                      // 0..23
  const int q0   = blockIdx.x * 128 + wave * 32;    // query group A base
  const int bidx = bh / NHEAD, h = bh % NHEAD;

  const unsigned short* QrowA = Qb + ((size_t)bh * NN + q0 + n16) * HD;
  const unsigned short* QrowB = QrowA + (size_t)16 * HD;
  const v16u qA0 = load_frag(QrowA, 0, hi),  qA1 = load_frag(QrowA, 32, hi);
  const v16u qB0 = load_frag(QrowB, 0, hi),  qB1 = load_frag(QrowB, 32, hi);

  const unsigned short* Kbase = Kb + (size_t)bh * NN * HD;
  const unsigned short* Vbase = Vt + (size_t)bh * HD * NN;
  const unsigned short* vrow0 = Vbase + (size_t)(0 * 16 + n16) * NN;
  const unsigned short* vrow1 = Vbase + (size_t)(1 * 16 + n16) * NN;
  const unsigned short* vrow2 = Vbase + (size_t)(2 * 16 + n16) * NN;
  const unsigned short* vrow3 = Vbase + (size_t)(3 * 16 + n16) * NN;

  v8f oA0 = {}, oA1 = {}, oA2 = {}, oA3 = {};
  v8f oB0 = {}, oB1 = {}, oB2 = {}, oB3 = {};
  float mA = -1e30f, lA = 0.0f, mB = -1e30f, lB = 0.0f;

  for (int kb = 0; kb < NN; kb += 32) {
    const unsigned short* krow0 = Kbase + (size_t)(kb + n16) * HD;
    const unsigned short* krow1 = Kbase + (size_t)(kb + 16 + n16) * HD;

    // K fragments for this 32-key block (shared by both query groups)
    v16u kf00 = load_frag(krow0, 0, hi), kf01 = load_frag(krow0, 32, hi);
    v16u kf10 = load_frag(krow1, 0, hi), kf11 = load_frag(krow1, 32, hi);
    // V fragments issued early so softmax VALU overlaps their latency
    v16u vf0 = load_frag(vrow0, kb, hi);
    v16u vf1 = load_frag(vrow1, kb, hi);
    v16u vf2 = load_frag(vrow2, kb, hi);
    v16u vf3 = load_frag(vrow3, kb, hi);

    v8f sA0 = {}, sA1 = {}, sB0 = {}, sB1 = {};
    sA0 = wmma_bf16(kf00, qA0, sA0); sA0 = wmma_bf16(kf01, qA1, sA0);
    sB0 = wmma_bf16(kf00, qB0, sB0); sB0 = wmma_bf16(kf01, qB1, sB0);
    sA1 = wmma_bf16(kf10, qA0, sA1); sA1 = wmma_bf16(kf11, qA1, sA1);
    sB1 = wmma_bf16(kf10, qB0, sB1); sB1 = wmma_bf16(kf11, qB1, sB1);

    // ---- online softmax, group A ----
    float fA;
    v16u pbA;
    {
      float mx = sA0[0];
#pragma unroll
      for (int r = 0; r < 8; ++r) { mx = fmaxf(mx, sA0[r]); mx = fmaxf(mx, sA1[r]); }
      mx = fmaxf(mx, __shfl_xor(mx, 16, 32));
      const float m_new = fmaxf(mA, mx);
      fA = __expf(mA - m_new);
      float sum = 0.0f;
#pragma unroll
      for (int r = 0; r < 8; ++r) {
        const float p0 = __expf(sA0[r] - m_new);
        const float p1 = __expf(sA1[r] - m_new);
        pbA[r] = f2bf(p0); pbA[8 + r] = f2bf(p1);
        sum += p0 + p1;
      }
      sum += __shfl_xor(sum, 16, 32);
      lA = lA * fA + sum;
      mA = m_new;
    }
    // ---- online softmax, group B ----
    float fB;
    v16u pbB;
    {
      float mx = sB0[0];
#pragma unroll
      for (int r = 0; r < 8; ++r) { mx = fmaxf(mx, sB0[r]); mx = fmaxf(mx, sB1[r]); }
      mx = fmaxf(mx, __shfl_xor(mx, 16, 32));
      const float m_new = fmaxf(mB, mx);
      fB = __expf(mB - m_new);
      float sum = 0.0f;
#pragma unroll
      for (int r = 0; r < 8; ++r) {
        const float p0 = __expf(sB0[r] - m_new);
        const float p1 = __expf(sB1[r] - m_new);
        pbB[r] = f2bf(p0); pbB[8 + r] = f2bf(p1);
        sum += p0 + p1;
      }
      sum += __shfl_xor(sum, 16, 32);
      lB = lB * fB + sum;
      mB = m_new;
    }

#pragma unroll
    for (int r = 0; r < 8; ++r) {
      oA0[r] *= fA; oA1[r] *= fA; oA2[r] *= fA; oA3[r] *= fA;
      oB0[r] *= fB; oB1[r] *= fB; oB2[r] *= fB; oB3[r] *= fB;
    }

    oA0 = wmma_bf16(vf0, pbA, oA0); oB0 = wmma_bf16(vf0, pbB, oB0);
    oA1 = wmma_bf16(vf1, pbA, oA1); oB1 = wmma_bf16(vf1, pbB, oB1);
    oA2 = wmma_bf16(vf2, pbA, oA2); oB2 = wmma_bf16(vf2, pbB, oB2);
    oA3 = wmma_bf16(vf3, pbA, oA3); oB3 = wmma_bf16(vf3, pbB, oB3);
  }

  const float rlA = 1.0f / lA;
  const float rlB = 1.0f / lB;
  unsigned short* orowA =
      Ob + (size_t)(bidx * NN + q0 + n16) * DIMC + h * HD;      // lane = query
  unsigned short* orowB = orowA + (size_t)16 * DIMC;
#pragma unroll
  for (int dt = 0; dt < 4; ++dt) {
    v8f oA = (dt == 0) ? oA0 : (dt == 1) ? oA1 : (dt == 2) ? oA2 : oA3;
    v8f oB = (dt == 0) ? oB0 : (dt == 1) ? oB1 : (dt == 2) ? oB2 : oB3;
#pragma unroll
    for (int j = 0; j < 4; ++j) {   // d = dt*16 + 8*hi + {2j, 2j+1}
      unsigned int pkA = (unsigned int)f2bf(oA[2 * j] * rlA) |
                         ((unsigned int)f2bf(oA[2 * j + 1] * rlA) << 16);
      unsigned int pkB = (unsigned int)f2bf(oB[2 * j] * rlB) |
                         ((unsigned int)f2bf(oB[2 * j + 1] * rlB) << 16);
      *(unsigned int*)(orowA + dt * 16 + 8 * hi + 2 * j) = pkA;
      *(unsigned int*)(orowB + dt * 16 + 8 * hi + 2 * j) = pkB;
    }
  }
}

// ---------------------------------------------------------------------------
// Output projection: [8192x768] @ proj_w^T + bias -> fp32 out.
// Same 32Mx64N wave tile + register double buffering as qkv_gemm_kernel.
__global__ __launch_bounds__(128) void proj_gemm_kernel(
    const unsigned short* __restrict__ ab,   // [8192 x 768] bf16
    const unsigned short* __restrict__ wb,   // [768 x 768] bf16
    const float* __restrict__ bias,          // [768]
    float* __restrict__ out) {               // [8192 x 768] fp32
  const int lane = threadIdx.x & 31;
  const int wave = threadIdx.x >> 5;
  const int n16  = lane & 15;
  const int hi   = lane >> 4;
  const int m0   = blockIdx.x * 128 + wave * 32;
  const int c0   = blockIdx.y * 64;

  const unsigned short* arowA = ab + (size_t)(m0 + n16) * DIMC;
  const unsigned short* arowB = ab + (size_t)(m0 + 16 + n16) * DIMC;
  const unsigned short* brow0 = wb + (size_t)(c0 + 0 * 16 + n16) * DIMC;
  const unsigned short* brow1 = wb + (size_t)(c0 + 1 * 16 + n16) * DIMC;
  const unsigned short* brow2 = wb + (size_t)(c0 + 2 * 16 + n16) * DIMC;
  const unsigned short* brow3 = wb + (size_t)(c0 + 3 * 16 + n16) * DIMC;

  v8f ac[2][4] = {};

  v16u aA = load_frag(arowA, 0, hi), aB = load_frag(arowB, 0, hi);
  v16u b0 = load_frag(brow0, 0, hi), b1 = load_frag(brow1, 0, hi);
  v16u b2 = load_frag(brow2, 0, hi), b3 = load_frag(brow3, 0, hi);

  for (int k = 0; k < DIMC; k += 32) {
    const int kn = (k + 32 < DIMC) ? k + 32 : 0;
    v16u aAn = load_frag(arowA, kn, hi), aBn = load_frag(arowB, kn, hi);
    v16u b0n = load_frag(brow0, kn, hi), b1n = load_frag(brow1, kn, hi);
    v16u b2n = load_frag(brow2, kn, hi), b3n = load_frag(brow3, kn, hi);

    ac[0][0] = wmma_bf16(aA, b0, ac[0][0]);
    ac[0][1] = wmma_bf16(aA, b1, ac[0][1]);
    ac[0][2] = wmma_bf16(aA, b2, ac[0][2]);
    ac[0][3] = wmma_bf16(aA, b3, ac[0][3]);
    ac[1][0] = wmma_bf16(aB, b0, ac[1][0]);
    ac[1][1] = wmma_bf16(aB, b1, ac[1][1]);
    ac[1][2] = wmma_bf16(aB, b2, ac[1][2]);
    ac[1][3] = wmma_bf16(aB, b3, ac[1][3]);

    aA = aAn; aB = aBn;
    b0 = b0n; b1 = b1n; b2 = b2n; b3 = b3n;
  }

#pragma unroll
  for (int g = 0; g < 2; ++g) {
#pragma unroll
    for (int j = 0; j < 4; ++j) {
      const v8f a = ac[g][j];
      const int col = c0 + j * 16 + n16;
      const float bia = bias[col];
#pragma unroll
      for (int r = 0; r < 8; ++r) {
        const int row = m0 + g * 16 + r + 8 * hi;
        out[(size_t)row * DIMC + col] = a[r] + bia;
      }
    }
  }
}

// ---------------------------------------------------------------------------
extern "C" void kernel_launch(void* const* d_in, const int* in_sizes, int n_in,
                              void* d_out, int out_size, void* d_ws,
                              size_t ws_size, hipStream_t stream) {
  (void)in_sizes; (void)n_in; (void)out_size; (void)ws_size;

  const float* x      = (const float*)d_in[0];  // [2,4096,768]
  const float* qkv_w  = (const float*)d_in[1];  // [2304,768]
  const float* qkv_b  = (const float*)d_in[2];  // [2304]
  const float* proj_w = (const float*)d_in[3];  // [768,768]
  const float* proj_b = (const float*)d_in[4];  // [768]
  float* out = (float*)d_out;                   // [2,4096,768]

  // Workspace carve-up (elements of u16/bf16); total ~67.6 MB.
  unsigned short* ws = (unsigned short*)d_ws;
  const size_t NXB   = (size_t)MTOK * DIMC;        // 6,291,456
  const size_t NWQKV = (size_t)3 * DIMC * DIMC;    // 1,769,472
  const size_t NWP   = (size_t)DIMC * DIMC;        //   589,824
  const size_t NQKV1 = (size_t)BB * NHEAD * NN * HD;  // == NXB

  unsigned short* xb    = ws;                size_t off = NXB;
  unsigned short* wqkv  = ws + off;          off += NWQKV;
  unsigned short* wproj = ws + off;          off += NWP;
  unsigned short* Qb    = ws + off;          off += NQKV1;
  unsigned short* Kb    = ws + off;          off += NQKV1;
  unsigned short* Vt    = ws + off;          off += NQKV1;
  unsigned short* attnb = ws + off;          off += NXB;

  f32_to_bf16_kernel<<<(int)((NXB + 255) / 256), 256, 0, stream>>>(x, xb, (int)NXB);
  f32_to_bf16_kernel<<<(int)((NWQKV + 255) / 256), 256, 0, stream>>>(qkv_w, wqkv, (int)NWQKV);
  f32_to_bf16_kernel<<<(int)((NWP + 255) / 256), 256, 0, stream>>>(proj_w, wproj, (int)NWP);

  qkv_gemm_kernel<<<dim3(MTOK / 128, (3 * DIMC) / 64), 128, 0, stream>>>(
      xb, wqkv, qkv_b, Qb, Kb, Vt);

  attn_kernel<<<dim3(NN / 128, BB * NHEAD), 128, 0, stream>>>(Qb, Kb, Vt, attnb);

  proj_gemm_kernel<<<dim3(MTOK / 128, DIMC / 64), 128, 0, stream>>>(
      attnb, wproj, proj_b, out);
}
